// MinGRU_90125593739639
// MI455X (gfx1250) — matrix operand order repaired and checked by reading, not compiled, not run on hardware
//
#include <hip/hip_runtime.h>
#include <hip/hip_bf16.h>

typedef __attribute__((ext_vector_type(16))) _Float16 v16h;
typedef __attribute__((ext_vector_type(8)))  _Float16 v8h;
typedef __attribute__((ext_vector_type(8)))  float    v8f;
typedef __attribute__((ext_vector_type(4)))  float    v4f;

#define BDIM 8
#define SDIM 4096
#define HDIM 512
#define KDIM 512
#define MTOT (BDIM * SDIM)      // 32768 rows

#define MBLK 128
#define NBLK 64
#define KSTEP 32
#define NKSTEPS (KDIM / KSTEP)  // 16
#define APAD 40                 // halves per LDS row (32 data + 8 pad)
#define STAGEH ((MBLK + 2 * NBLK) * APAD)   // 10240 halves = 20 KB per stage

#define CCH  32                 // scan chunks
#define CLEN (SDIM / CCH)       // 128 steps per chunk

// ---------- async global->LDS copy of 32 contiguous bytes (16 halves) ------
__device__ inline void async_cp16h(const _Float16* g, _Float16* l) {
    unsigned lds = (unsigned)(unsigned long long)(void*)l;   // low 32 bits = LDS byte offset
    asm volatile(
        "global_load_async_to_lds_b128 %0, %1, off\n\t"
        "global_load_async_to_lds_b128 %0, %1, off offset:16"
        :: "v"(lds), "v"(g)
        : "memory");
}
__device__ inline void wait_async(int keep) {
    if (keep) asm volatile("s_wait_asynccnt 0x4" ::: "memory");
    else      asm volatile("s_wait_asynccnt 0x0" ::: "memory");
}

// ---------- WMMA fragment gathers (ISA 16-bit A 16x32 / B 32x16 layouts) ----
__device__ inline v16h frag_a(const _Float16* base, int lane) {
    // lane holds row m = lane&15; lanes>=16 take K-halves offset by 8
    int m  = lane & 15;
    int kh = lane >> 4;
    const _Float16* p = base + m * APAD;
    v8h c0 = *(const v8h*)(p + kh * 8);          // K = kh*8 .. +7
    v8h c1 = *(const v8h*)(p + 16 + kh * 8);     // K = 16 + kh*8 .. +7
    v16h f;
#pragma unroll
    for (int i = 0; i < 8; ++i) { f[i] = c0[i]; f[i + 8] = c1[i]; }
    return f;
}
__device__ inline v16h frag_b(const _Float16* base, int lane) {
    // lane holds column n = lane&15 of B (== row n of W), K = kb..kb+7 and 16+kb..16+kb+7, kb = (lane>>4)*8
    int n  = lane & 15;
    int kb = (lane >> 4) * 8;
    const _Float16* p = base + n * APAD;
    v8h c0 = *(const v8h*)(p + kb);
    v8h c1 = *(const v8h*)(p + 16 + kb);
    v16h f;
#pragma unroll
    for (int i = 0; i < 8; ++i) { f[i] = c0[i]; f[i + 8] = c1[i]; }
    return f;
}

// ---------- fp32 -> f16 pre-conversion -------------------------------------
__global__ __launch_bounds__(256) void cvt_f16_kernel(
    const float* __restrict__ in, _Float16* __restrict__ out)
{
    const size_t i = ((size_t)blockIdx.x * blockDim.x + threadIdx.x) * 8;
    v4f f0 = *(const v4f*)(in + i);
    v4f f1 = *(const v4f*)(in + i + 4);
    v8h h;
#pragma unroll
    for (int j = 0; j < 4; ++j) { h[j] = (_Float16)f0[j]; h[j + 4] = (_Float16)f1[j]; }
    *(v8h*)(out + i) = h;
}

// ---------- fused dual-GEMM + gate epilogue --------------------------------
// k = X*wz^T + bz, hx = X*wh^T + bh for a 128x64 tile, then
// a = sigmoid(-k), b = sigmoid(k)*g(hx), packed as {f16 a, f16 b} in u32.
__global__ __launch_bounds__(256) void gemm_gate_kernel(
    const _Float16* __restrict__ X,
    const _Float16* __restrict__ wz, const float* __restrict__ bz,
    const _Float16* __restrict__ wh, const float* __restrict__ bh,
    unsigned* __restrict__ AB)
{
    __shared__ _Float16 sMem[2][STAGEH];   // [stage][ A | Bz | Bh ]

    const int m0 = blockIdx.x * MBLK;
    const int n0 = blockIdx.y * NBLK;
    const int t  = threadIdx.x;
    const int lane = t & 31;
    const int wid  = t >> 5;
    const int wm = (wid & 3) * 32;   // 4 waves tile M
    const int wn = (wid >> 2) * 32;  // 2 waves tile N

    // per-thread copy map (4 async b128 per thread per stage)
    const int arow = t >> 1, acb = (t & 1) * 16;
    const int mat = t >> 7, tt = t & 127;
    const int wrow = tt >> 1, wcb = (tt & 1) * 16;
    const _Float16* wsel = mat ? wh : wz;
    const _Float16* gA = X + (size_t)(m0 + arow) * KDIM + acb;
    const _Float16* gW = wsel + (size_t)(n0 + wrow) * KDIM + wcb;
    const int lA = arow * APAD + acb;
    const int lW = MBLK * APAD + mat * (NBLK * APAD) + wrow * APAD + wcb;

    const v8f vzero = {0.f, 0.f, 0.f, 0.f, 0.f, 0.f, 0.f, 0.f};
    v8f aK[2][2], aH[2][2];
#pragma unroll
    for (int i = 0; i < 2; ++i)
#pragma unroll
        for (int j = 0; j < 2; ++j) { aK[i][j] = vzero; aH[i][j] = vzero; }

    // prologue: stage 0
    async_cp16h(gA, &sMem[0][lA]);
    async_cp16h(gW, &sMem[0][lW]);

    for (int ks = 0; ks < NKSTEPS; ++ks) {
        const int buf = ks & 1;
        const int more = (ks + 1 < NKSTEPS);
        if (more) {   // prefetch next stage while computing this one
            const int kn = (ks + 1) * KSTEP;
            async_cp16h(gA + kn, &sMem[buf ^ 1][lA]);
            async_cp16h(gW + kn, &sMem[buf ^ 1][lW]);
        }
        wait_async(more);          // oldest stage complete (in-order retire)
        __syncthreads();

        const _Float16* sA  = &sMem[buf][0];
        const _Float16* sBz = &sMem[buf][MBLK * APAD];
        const _Float16* sBh = &sMem[buf][MBLK * APAD + NBLK * APAD];

        // gather all fragments first, then issue WMMAs back-to-back
        v16h fa0  = frag_a(sA + (wm + 0)  * APAD, lane);
        v16h fa1  = frag_a(sA + (wm + 16) * APAD, lane);
        v16h fbz0 = frag_b(sBz + (wn + 0)  * APAD, lane);
        v16h fbz1 = frag_b(sBz + (wn + 16) * APAD, lane);
        v16h fbh0 = frag_b(sBh + (wn + 0)  * APAD, lane);
        v16h fbh1 = frag_b(sBh + (wn + 16) * APAD, lane);

        aK[0][0] = __builtin_amdgcn_wmma_f32_16x16x32_f16(false, fa0, false, fbz0, (short)0, aK[0][0], false, false);
        aK[1][0] = __builtin_amdgcn_wmma_f32_16x16x32_f16(false, fa1, false, fbz0, (short)0, aK[1][0], false, false);
        aK[0][1] = __builtin_amdgcn_wmma_f32_16x16x32_f16(false, fa0, false, fbz1, (short)0, aK[0][1], false, false);
        aK[1][1] = __builtin_amdgcn_wmma_f32_16x16x32_f16(false, fa1, false, fbz1, (short)0, aK[1][1], false, false);
        aH[0][0] = __builtin_amdgcn_wmma_f32_16x16x32_f16(false, fa0, false, fbh0, (short)0, aH[0][0], false, false);
        aH[1][0] = __builtin_amdgcn_wmma_f32_16x16x32_f16(false, fa1, false, fbh0, (short)0, aH[1][0], false, false);
        aH[0][1] = __builtin_amdgcn_wmma_f32_16x16x32_f16(false, fa0, false, fbh1, (short)0, aH[0][1], false, false);
        aH[1][1] = __builtin_amdgcn_wmma_f32_16x16x32_f16(false, fa1, false, fbh1, (short)0, aH[1][1], false, false);

        __syncthreads();           // all waves done with buf before it is reissued
    }

    // epilogue: C layout lane -> (m = 8*(lane>>4)+r, n = lane&15)
    const int lm = lane >> 4, ln = lane & 15;
#pragma unroll
    for (int nt = 0; nt < 2; ++nt) {
        const int n = n0 + wn + nt * 16 + ln;
        const float bzv = bz[n];
        const float bhv = bh[n];
#pragma unroll
        for (int mt = 0; mt < 2; ++mt) {
#pragma unroll
            for (int r = 0; r < 8; ++r) {
                const int m = m0 + wm + mt * 16 + lm * 8 + r;
                const float kv = aK[mt][nt][r] + bzv;
                const float hv = aH[mt][nt][r] + bhv;
                const float z  = 1.0f / (1.0f + __expf(-kv));     // sigmoid(k)
                const float ac = 1.0f - z;                        // sigmoid(-k) = exp(log_coeffs)
                const float ht = (hv >= 0.0f) ? (hv + 0.5f)
                                              : (1.0f / (1.0f + __expf(-hv)));   // g(hx)
                const float bv = z * ht;                          // exp(log_values)
                union { _Float16 h2[2]; unsigned u; } pk;
                pk.h2[0] = (_Float16)ac;
                pk.h2[1] = (_Float16)bv;
                AB[(size_t)m * HDIM + n] = pk.u;
            }
        }
    }
}

// ---------- chunked linear scan: h_t = a_t*h_{t-1} + b_t -------------------
__global__ __launch_bounds__(256) void scan_phase1(
    const unsigned* __restrict__ AB, float* __restrict__ PQ)
{
    const int tid = blockIdx.x * blockDim.x + threadIdx.x;   // B*H*CCH threads
    const int h = tid & (HDIM - 1);
    const int b = (tid >> 9) & (BDIM - 1);
    const int c = tid >> 12;
    float P = 1.0f, Q = 0.0f;
    size_t base = ((size_t)b * SDIM + (size_t)c * CLEN) * HDIM + h;
    for (int s = 0; s < CLEN; ++s) {
        union { unsigned u; _Float16 h2[2]; } pk;
        pk.u = AB[base + (size_t)s * HDIM];
        const float a = (float)pk.h2[0];
        const float v = (float)pk.h2[1];
        P *= a;
        Q = a * Q + v;
    }
    const size_t ci = (size_t)(b * HDIM + h) * CCH + c;
    PQ[2 * ci + 0] = P;
    PQ[2 * ci + 1] = Q;
}

__global__ __launch_bounds__(256) void scan_phase2(
    const float* __restrict__ PQ, float* __restrict__ HIN,
    float* __restrict__ fin)
{
    const int chn = blockIdx.x * blockDim.x + threadIdx.x;   // B*H threads
    float hcur = 0.5f;                                       // h0 = g(0) = 0.5
    const size_t base = (size_t)chn * CCH;
    for (int c = 0; c < CCH; ++c) {
        HIN[base + c] = hcur;
        hcur = PQ[2 * (base + c)] * hcur + PQ[2 * (base + c) + 1];
    }
    fin[chn] = hcur;                                         // final h (s = S-1)
}

template <typename OutT>
__global__ __launch_bounds__(256) void scan_phase3(
    const unsigned* __restrict__ AB, const float* __restrict__ HIN,
    OutT* __restrict__ out)
{
    const int tid = blockIdx.x * blockDim.x + threadIdx.x;
    const int h = tid & (HDIM - 1);
    const int b = (tid >> 9) & (BDIM - 1);
    const int c = tid >> 12;
    float hcur = HIN[(size_t)(b * HDIM + h) * CCH + c];
    size_t base = ((size_t)b * SDIM + (size_t)c * CLEN) * HDIM + h;
    for (int s = 0; s < CLEN; ++s) {
        union { unsigned u; _Float16 h2[2]; } pk;
        pk.u = AB[base + (size_t)s * HDIM];
        hcur = (float)pk.h2[0] * hcur + (float)pk.h2[1];
        out[base + (size_t)s * HDIM] = (OutT)hcur;
    }
}

// ---------- host-side orchestration ----------------------------------------
extern "C" void kernel_launch(void* const* d_in, const int* in_sizes, int n_in,
                              void* d_out, int out_size, void* d_ws, size_t ws_size,
                              hipStream_t stream)
{
    const float* x   = (const float*)d_in[0];
    const float* wz0 = (const float*)d_in[1];
    const float* bz0 = (const float*)d_in[2];
    const float* wh0 = (const float*)d_in[3];
    const float* bh0 = (const float*)d_in[4];
    const float* wz1 = (const float*)d_in[5];
    const float* bz1 = (const float*)d_in[6];
    const float* wh1 = (const float*)d_in[7];
    const float* bh1 = (const float*)d_in[8];
    float* out = (float*)d_out;

    // workspace layout
    char* ws = (char*)d_ws;
    unsigned*  AB   = (unsigned*)ws;                                    // 64 MiB
    _Float16*  x16  = (_Float16*)(ws + (size_t)MTOT * HDIM * 4);        // 32 MiB
    _Float16*  hmid = (_Float16*)(ws + (size_t)MTOT * HDIM * 6);        // 32 MiB
    _Float16*  w16  = (_Float16*)(ws + (size_t)MTOT * HDIM * 8);        // 4 x 0.5 MiB
    _Float16*  wz0h = w16;
    _Float16*  wh0h = w16 + (size_t)HDIM * KDIM;
    _Float16*  wz1h = w16 + (size_t)2 * HDIM * KDIM;
    _Float16*  wh1h = w16 + (size_t)3 * HDIM * KDIM;
    float*     PQ   = (float*)(ws + (size_t)MTOT * HDIM * 8 + (size_t)4 * HDIM * KDIM * 2);
    float*     HIN  = PQ + (size_t)2 * BDIM * HDIM * CCH;

    float* fin0 = out + (size_t)MTOT * HDIM;       // stacked finals: (L,B,1,H)
    float* fin1 = fin0 + BDIM * HDIM;

    dim3 ggrid(MTOT / MBLK, HDIM / NBLK);          // 256 x 8 blocks
    const int scan_blk = (BDIM * HDIM * CCH) / 256;
    const int ch_blk   = (BDIM * HDIM) / 256;
    const int xcvt_blk = (MTOT * HDIM / 8) / 256;  // 8192
    const int wcvt_blk = (HDIM * KDIM / 8) / 256;  // 128

    // one-time f16 conversions (x + all weights)
    cvt_f16_kernel<<<xcvt_blk, 256, 0, stream>>>(x,   x16);
    cvt_f16_kernel<<<wcvt_blk, 256, 0, stream>>>(wz0, wz0h);
    cvt_f16_kernel<<<wcvt_blk, 256, 0, stream>>>(wh0, wh0h);
    cvt_f16_kernel<<<wcvt_blk, 256, 0, stream>>>(wz1, wz1h);
    cvt_f16_kernel<<<wcvt_blk, 256, 0, stream>>>(wh1, wh1h);

    // Layer 0
    gemm_gate_kernel<<<ggrid, 256, 0, stream>>>(x16, wz0h, bz0, wh0h, bh0, AB);
    scan_phase1<<<scan_blk, 256, 0, stream>>>(AB, PQ);
    scan_phase2<<<ch_blk, 256, 0, stream>>>(PQ, HIN, fin0);
    scan_phase3<_Float16><<<scan_blk, 256, 0, stream>>>(AB, HIN, hmid);

    // Layer 1 (f16 hidden state feeds straight back into WMMA)
    gemm_gate_kernel<<<ggrid, 256, 0, stream>>>(hmid, wz1h, bz1, wh1h, bh1, AB);
    scan_phase1<<<scan_blk, 256, 0, stream>>>(AB, PQ);
    scan_phase2<<<ch_blk, 256, 0, stream>>>(PQ, HIN, fin1);
    scan_phase3<float><<<scan_blk, 256, 0, stream>>>(AB, HIN, out);
}
